// BrainGAT_54597624267076
// MI455X (gfx1250) — matrix operand order repaired and checked by reading, not compiled, not run on hardware
//
#include <hip/hip_runtime.h>
#include <hip/hip_bf16.h>
#include <cstdint>

// ---------------------------------------------------------------------------
// Types for CDNA5 WMMA (wave32): A/B = 16 bf16 per lane, C/D = 8 f32 per lane
// ---------------------------------------------------------------------------
typedef __attribute__((ext_vector_type(16))) __bf16 v16bf;
typedef __attribute__((ext_vector_type(8)))  float  v8f;

// ---------------------------------------------------------------------------
// Order-preserving float <-> uint mapping for atomic scatter-max
// ---------------------------------------------------------------------------
__device__ __forceinline__ unsigned encf(float f) {
    unsigned u = __float_as_uint(f);
    return (u & 0x80000000u) ? ~u : (u | 0x80000000u);
}
__device__ __forceinline__ float decf(unsigned u) {
    u = (u & 0x80000000u) ? (u & 0x7FFFFFFFu) : ~u;
    return __uint_as_float(u);
}

// ---------------------------------------------------------------------------
// Pack A[M,K] (f32, row-major) into per-wave WMMA A-fragment order (bf16).
// Packed element t = ((mtile*KC + kc)*32 + lane)*16 + j, where fragment
// VGPR v=j>>1 holds K pair ko=(v<4 ? 2v : 8+2v) + (lane>=16 ? 8 : 0).
// ---------------------------------------------------------------------------
__global__ void __launch_bounds__(256)
k_pack_a(const float* __restrict__ A, __bf16* __restrict__ Ap, int K, int total) {
    const int t = blockIdx.x * blockDim.x + threadIdx.x;
    if (t >= total) return;
    const int j    = t & 15;
    const int lane = (t >> 4) & 31;
    const int rest = t >> 9;
    const int KC   = K >> 5;
    const int kc   = rest % KC;
    const int tile = rest / KC;
    const int row  = tile * 16 + (lane & 15);
    const int kb   = (lane >> 4) * 8;
    const int v    = j >> 1;
    const int k    = kc * 32 + ((v < 4) ? 2 * v : 8 + 2 * v) + kb + (j & 1);
    Ap[t] = (__bf16)A[(size_t)row * K + k];
}

// ---------------------------------------------------------------------------
// Pack B[K,N] (f32, row-major) into per-wave WMMA B-fragment order (bf16).
// B 32x16: lanes 0-15 hold K=0..7,16..23, lanes 16-31 hold K=8..15,24..31, N = lane&15.
// Packed element t = ((ntile*KC + kc)*32 + lane)*16 + j.
// ---------------------------------------------------------------------------
__global__ void __launch_bounds__(256)
k_pack_b(const float* __restrict__ B, __bf16* __restrict__ Bp,
         int K, int N, int total) {
    const int t = blockIdx.x * blockDim.x + threadIdx.x;
    if (t >= total) return;
    const int j     = t & 15;
    const int lane  = (t >> 4) & 31;
    const int rest  = t >> 9;
    const int KC    = K >> 5;
    const int kc    = rest % KC;
    const int ntile = rest / KC;
    const int n     = ntile * 16 + (lane & 15);
    const int kb    = (lane >> 4) * 8;
    const int k     = kc * 32 + kb + j + (j & 8);
    Bp[t] = (__bf16)B[(size_t)k * N + n];
}

// ---------------------------------------------------------------------------
// WMMA GEMM on pre-packed fragments: C[M,N] = A@B (+bias).
// Block = 128 threads = 4 waves; each wave computes two adjacent 16x16 C
// tiles sharing one A fragment: load {a, b0, b1} in one clause -> wmma, wmma.
// No LDS, no barriers, no divergence in the hot loop.
// REQUIRES: M%16==0, K%32==0, N%32==0 (every wave owns a full tile pair).
// ---------------------------------------------------------------------------
__global__ void __launch_bounds__(128)
k_gemm_wmma(const __bf16* __restrict__ Ap, const __bf16* __restrict__ Bp,
            const float* __restrict__ bias, float* __restrict__ C,
            int M, int K, int N, int addBias) {
    const int lane  = threadIdx.x & 31;
    const int wave  = threadIdx.x >> 5;
    const int mtile = blockIdx.x;
    const int nt0   = (blockIdx.y * 4 + wave) * 2;
    const int KC    = K >> 5;
    if (nt0 * 16 >= N) return;                 // wave-uniform

    const v16bf* af  = (const v16bf*)Ap + (size_t)mtile * KC * 32 + lane;
    const v16bf* bf0 = (const v16bf*)Bp + (size_t)nt0 * KC * 32 + lane;
    const v16bf* bf1 = (const v16bf*)Bp + (size_t)(nt0 + 1) * KC * 32 + lane;

    v8f acc0 = {};
    v8f acc1 = {};
    for (int kc = 0; kc < KC; ++kc) {
        const v16bf a  = af [(size_t)kc * 32];
        const v16bf b0 = bf0[(size_t)kc * 32];
        const v16bf b1 = bf1[(size_t)kc * 32];
        acc0 = __builtin_amdgcn_wmma_f32_16x16x32_bf16(
                   false, a, false, b0, (short)0, acc0, false, false);
        acc1 = __builtin_amdgcn_wmma_f32_16x16x32_bf16(
                   false, a, false, b1, (short)0, acc1, false, false);
    }

    const int nidx = lane & 15;
    const int hi   = lane >> 4;
    {
        const int col = nt0 * 16 + nidx;
        const float bv = addBias ? bias[col] : 0.0f;
        #pragma unroll
        for (int r = 0; r < 8; ++r)
            C[(size_t)(mtile * 16 + r + hi * 8) * N + col] = acc0[r] + bv;
    }
    {
        const int col = (nt0 + 1) * 16 + nidx;
        const float bv = addBias ? bias[col] : 0.0f;
        #pragma unroll
        for (int r = 0; r < 8; ++r)
            C[(size_t)(mtile * 16 + r + hi * 8) * N + col] = acc1[r] + bv;
    }
}

// ---------------------------------------------------------------------------
// Edge pass 1: logits[e,h] = att[h,:] . leaky_relu(xl[src]+xr[dst]); atomic
// scatter-max into lmaxEnc[dst,h].  One wave (32 lanes) per edge.
// ---------------------------------------------------------------------------
__global__ void __launch_bounds__(256)
k_edge_logits(const int* __restrict__ ei, int E0, int Etot,
              const float* __restrict__ xl, const float* __restrict__ xr,
              const float* __restrict__ att, float* __restrict__ logits,
              unsigned* __restrict__ lmaxEnc, int H, int C) {
    const int e = blockIdx.x * 8 + (threadIdx.x >> 5);
    if (e >= Etot) return;
    const int lane = threadIdx.x & 31;
    int s, d;
    if (e < E0) { s = ei[e]; d = ei[E0 + e]; } else { s = e - E0; d = s; }

    const int HC = H * C;
    const float* xs = xl + (size_t)s * HC;
    const float* xd = xr + (size_t)d * HC;
    __builtin_prefetch(xs + lane * 8, 0, 0);   // global_prefetch_b8
    __builtin_prefetch(xd + lane * 8, 0, 0);

    for (int h = 0; h < H; ++h) {
        float acc = 0.f;
        for (int c = lane; c < C; c += 32) {
            float v = xs[h * C + c] + xd[h * C + c];
            v = v > 0.f ? v : 0.2f * v;
            acc += v * att[h * C + c];
        }
        #pragma unroll
        for (int off = 16; off > 0; off >>= 1)
            acc += __shfl_xor(acc, off, 32);
        if (lane == 0) {
            logits[(size_t)e * H + h] = acc;
            atomicMax(&lmaxEnc[(size_t)d * H + h], encf(acc));
        }
    }
}

// ---------------------------------------------------------------------------
// Edge pass 2: p = exp(logit - max[dst]); denom[dst,h] += p.  Thread per (e,h).
// ---------------------------------------------------------------------------
__global__ void __launch_bounds__(256)
k_edge_softmax_p(const int* __restrict__ ei, int E0, int Etot,
                 float* __restrict__ logits, const unsigned* __restrict__ lmaxEnc,
                 float* __restrict__ denom, int H) {
    const int t = blockIdx.x * blockDim.x + threadIdx.x;
    if (t >= Etot * H) return;
    const int e = t / H;
    const int h = t - e * H;
    const int d = (e < E0) ? ei[E0 + e] : (e - E0);
    const float m = decf(lmaxEnc[(size_t)d * H + h]);
    const float p = __expf(logits[t] - m);
    logits[t] = p;
    atomicAdd(&denom[(size_t)d * H + h], p);
}

// ---------------------------------------------------------------------------
// Edge pass 3: agg[dst] += alpha * xl[src].  One wave per edge, lane strides
// the H*C channels; f32 atomics resolve in L2.
// ---------------------------------------------------------------------------
__global__ void __launch_bounds__(256)
k_edge_aggregate(const int* __restrict__ ei, int E0, int Etot,
                 const float* __restrict__ xl, const float* __restrict__ p,
                 const float* __restrict__ denom, float* __restrict__ agg,
                 int H, int C, int logC) {
    const int e = blockIdx.x * 8 + (threadIdx.x >> 5);
    if (e >= Etot) return;
    const int lane = threadIdx.x & 31;
    int s, d;
    if (e < E0) { s = ei[e]; d = ei[E0 + e]; } else { s = e - E0; d = s; }

    const int HC = H * C;
    float alpha[4];
    #pragma unroll
    for (int h = 0; h < 4; ++h)
        alpha[h] = p[(size_t)e * 4 + h] / (denom[(size_t)d * 4 + h] + 1e-16f);

    const float* xs = xl + (size_t)s * HC;
    float* ag = agg + (size_t)d * HC;
    __builtin_prefetch(xs + lane * 8, 0, 0);
    for (int c = lane; c < HC; c += 32)
        atomicAdd(&ag[c], alpha[c >> logC] * xs[c]);
}

// ---------------------------------------------------------------------------
// out[i] = elu(out[i] + bias[i & (HC-1)])   (HC is a power of two)
// ---------------------------------------------------------------------------
__global__ void __launch_bounds__(256)
k_bias_elu(float* __restrict__ h, const float* __restrict__ b, int n, int HCm1) {
    const int i = blockIdx.x * blockDim.x + threadIdx.x;
    if (i >= n) return;
    float v = h[i] + b[i & HCm1];
    h[i] = v > 0.f ? v : (__expf(v) - 1.f);
}

// ---------------------------------------------------------------------------
// Launch
// ---------------------------------------------------------------------------
extern "C" void kernel_launch(void* const* d_in, const int* in_sizes, int n_in,
                              void* d_out, int out_size, void* d_ws, size_t ws_size,
                              hipStream_t stream) {
    (void)n_in; (void)out_size; (void)ws_size;
    constexpr int IN_CH = 128, H = 4, C1 = 64, C2 = 32, OUT_CH = 64;
    constexpr int HC1 = H * C1;   // 256
    constexpr int HC2 = H * C2;   // 128

    const float* x    = (const float*)d_in[0];
    const int*   ei   = (const int*)  d_in[1];
    const float* Wl1  = (const float*)d_in[2];
    const float* Wr1  = (const float*)d_in[3];
    const float* att1 = (const float*)d_in[4];
    const float* b1   = (const float*)d_in[5];
    const float* Wl2  = (const float*)d_in[6];
    const float* Wr2  = (const float*)d_in[7];
    const float* att2 = (const float*)d_in[8];
    const float* b2   = (const float*)d_in[9];
    const float* Wfc  = (const float*)d_in[10];
    const float* bfc  = (const float*)d_in[11];
    float* out = (float*)d_out;

    const int NN   = in_sizes[0] / IN_CH;     // 50000
    const int E0   = in_sizes[1] / 2;         // 800000
    const int Etot = E0 + NN;                 // + self loops

    // ---- carve workspace -------------------------------------------------
    char* w = (char*)d_ws;
    auto carve = [&](size_t bytes) -> void* {
        void* p = (void*)w;
        w += (bytes + 255) & ~(size_t)255;
        return p;
    };
    __bf16* Wl1p = (__bf16*)carve(sizeof(__bf16) * IN_CH * HC1);
    __bf16* Wr1p = (__bf16*)carve(sizeof(__bf16) * IN_CH * HC1);
    __bf16* Wl2p = (__bf16*)carve(sizeof(__bf16) * HC1 * HC2);
    __bf16* Wr2p = (__bf16*)carve(sizeof(__bf16) * HC1 * HC2);
    __bf16* Wfcp = (__bf16*)carve(sizeof(__bf16) * HC2 * OUT_CH);
    __bf16* Ap   = (__bf16*)carve(sizeof(__bf16) * (size_t)NN * HC1); // max M*K
    float* xl1   = (float*)carve(sizeof(float) * (size_t)NN * HC1);
    float* xr1   = (float*)carve(sizeof(float) * (size_t)NN * HC1);
    float* h1    = (float*)carve(sizeof(float) * (size_t)NN * HC1);
    float* xl2   = (float*)carve(sizeof(float) * (size_t)NN * HC2);
    float* xr2   = (float*)carve(sizeof(float) * (size_t)NN * HC2);
    float* h2    = (float*)carve(sizeof(float) * (size_t)NN * HC2);
    float* logits = (float*)carve(sizeof(float) * (size_t)Etot * H);
    unsigned* lmax = (unsigned*)carve(sizeof(unsigned) * (size_t)NN * H);
    float* denom   = (float*)carve(sizeof(float) * (size_t)NN * H);

    auto packB = [&](const float* s, __bf16* dst, int K, int N) {
        const int total = K * N;
        k_pack_b<<<(total + 255) / 256, 256, 0, stream>>>(s, dst, K, N, total);
    };
    packB(Wl1, Wl1p, IN_CH, HC1);
    packB(Wr1, Wr1p, IN_CH, HC1);
    packB(Wl2, Wl2p, HC1, HC2);
    packB(Wr2, Wr2p, HC1, HC2);
    packB(Wfc, Wfcp, HC2, OUT_CH);

    auto packA = [&](const float* s, int K) {
        const int total = NN * K;
        k_pack_a<<<(total + 255) / 256, 256, 0, stream>>>(s, Ap, K, total);
    };
    auto gemm = [&](const __bf16* Bp, const float* bias, float* Cp,
                    int K, int N, int addBias) {
        dim3 grid(NN / 16, (N + 127) / 128);
        k_gemm_wmma<<<grid, 128, 0, stream>>>(Ap, Bp, bias, Cp, NN, K, N, addBias);
    };

    const int eBlocks  = (Etot + 7) / 8;
    const int ehBlocks = (Etot * H + 255) / 256;

    // ---- layer 1 ---------------------------------------------------------
    packA(x, IN_CH);
    gemm(Wl1p, nullptr, xl1, IN_CH, HC1, 0);
    gemm(Wr1p, nullptr, xr1, IN_CH, HC1, 0);
    hipMemsetAsync(h1, 0, sizeof(float) * (size_t)NN * HC1, stream);
    hipMemsetAsync(lmax, 0, sizeof(unsigned) * (size_t)NN * H, stream);
    hipMemsetAsync(denom, 0, sizeof(float) * (size_t)NN * H, stream);
    k_edge_logits<<<eBlocks, 256, 0, stream>>>(ei, E0, Etot, xl1, xr1, att1,
                                               logits, lmax, H, C1);
    k_edge_softmax_p<<<ehBlocks, 256, 0, stream>>>(ei, E0, Etot, logits, lmax,
                                                   denom, H);
    k_edge_aggregate<<<eBlocks, 256, 0, stream>>>(ei, E0, Etot, xl1, logits,
                                                  denom, h1, H, C1, 6);
    k_bias_elu<<<(NN * HC1 + 255) / 256, 256, 0, stream>>>(h1, b1, NN * HC1, HC1 - 1);

    // ---- layer 2 ---------------------------------------------------------
    packA(h1, HC1);
    gemm(Wl2p, nullptr, xl2, HC1, HC2, 0);
    gemm(Wr2p, nullptr, xr2, HC1, HC2, 0);
    hipMemsetAsync(h2, 0, sizeof(float) * (size_t)NN * HC2, stream);
    hipMemsetAsync(lmax, 0, sizeof(unsigned) * (size_t)NN * H, stream);
    hipMemsetAsync(denom, 0, sizeof(float) * (size_t)NN * H, stream);
    k_edge_logits<<<eBlocks, 256, 0, stream>>>(ei, E0, Etot, xl2, xr2, att2,
                                               logits, lmax, H, C2);
    k_edge_softmax_p<<<ehBlocks, 256, 0, stream>>>(ei, E0, Etot, logits, lmax,
                                                   denom, H);
    k_edge_aggregate<<<eBlocks, 256, 0, stream>>>(ei, E0, Etot, xl2, logits,
                                                  denom, h2, H, C2, 5);
    k_bias_elu<<<(NN * HC2 + 255) / 256, 256, 0, stream>>>(h2, b2, NN * HC2, HC2 - 1);

    // ---- final linear (bias fused, no activation) ------------------------
    packA(h2, HC2);
    gemm(Wfcp, bfc, out, HC2, OUT_CH, 1);
}
